// VectorQuantizerEMA_76587856823007
// MI455X (gfx1250) — matrix operand order repaired and checked by reading, not compiled, not run on hardware
//
#include <hip/hip_runtime.h>
#include <hip/hip_bf16.h>

typedef __attribute__((ext_vector_type(16))) __bf16 v16bf;
typedef __attribute__((ext_vector_type(8)))  float  v8f;
typedef __attribute__((ext_vector_type(4)))  unsigned int u32x4;
typedef __attribute__((ext_vector_type(8)))  int i32x8;
typedef __attribute__((ext_vector_type(4)))  int i32x4;

#define N_ROWS  65536
#define DIM     64
#define KCODES  512
#define NTILES  32        // 512 / 16 column tiles
#define GAMMA_  0.99f
#define BETA_   0.25f
#define EPS_    1e-5f

// ---- workspace layout (bytes) ----
#define WS_CBFRAG 0        // 32768 bf16 = 65536 B  (B-fragment swizzled codebook)
#define WS_CNORM  65536    // 512 f32               (||c_k||^2)
#define WS_HIST   67584    // 512 f32               (cluster size accumulator)
#define WS_LOSS   69632    // 1 f32 (padded)        (sum of squared errors)
#define WS_DWACC  69888    // 32768 f32             (dw accumulator, D x K)

// ---- output layout (f32 elements) ----
#define O_STE   0
#define O_PERP  4194304
#define O_LOSS  4194305
#define O_CB    4194306
#define O_CS    4227074
#define O_DW    4227586

#if __has_builtin(__builtin_amdgcn_tensor_load_to_lds)
#define USE_TDM 1
#else
#define USE_TDM 0
#endif

// -----------------------------------------------------------------------------
// Prep: swizzle codebook into WMMA B-fragment order (bf16), column norms,
// zero the per-launch accumulators.
// B-fragment layout (16-bit B, 32x16): lane n (0..15) holds K=0..7,16..23 in elements
// 0..15; lane n+16 holds K=8..15,24..31.  Fragment id = tile*2 + kb (kb = K-block of 32).
// -----------------------------------------------------------------------------
__global__ __launch_bounds__(256) void vq_prep(const float* __restrict__ cb,
                                               char* __restrict__ ws) {
    int tid = blockIdx.x * blockDim.x + threadIdx.x;
    __bf16* cbfrag = (__bf16*)(ws + WS_CBFRAG);
    float*  cnorm  = (float*)(ws + WS_CNORM);
    float*  hist   = (float*)(ws + WS_HIST);
    float*  lossa  = (float*)(ws + WS_LOSS);
    float*  dwacc  = (float*)(ws + WS_DWACC);

    if (tid < 2048) {                       // 32 tiles * 2 kb * 32 lanes
        int t    = tid >> 6;
        int kb   = (tid >> 5) & 1;
        int lane = tid & 31;
        int half = lane >> 4;
        int n    = lane & 15;
        int col  = t * 16 + n;
#pragma unroll
        for (int e = 0; e < 16; ++e) {
            int k = kb * 32 + (e & 8) * 2 + half * 8 + (e & 7);   // global contraction row
            cbfrag[tid * 16 + e] = (__bf16)cb[k * KCODES + col];
        }
    } else if (tid < 2560) {                // column norms
        int k = tid - 2048;
        float s = 0.f;
        for (int d = 0; d < DIM; ++d) { float v = cb[d * KCODES + k]; s += v * v; }
        cnorm[k] = s;
    } else if (tid < 3072) {
        hist[tid - 2560] = 0.f;
    } else if (tid == 3072) {
        lossa[0] = 0.f;
    } else if (tid >= 4096 && tid < 4096 + DIM * KCODES) {
        dwacc[tid - 4096] = 0.f;
    }
}

// -----------------------------------------------------------------------------
// Main: per-wave 2x16-row tiles x 2 column tiles per iteration
// -> 4 independent WMMA accumulation chains, 8 WMMAs / iteration, which covers
// the 5-slot bf16 WMMA RAW hazard window without v_nops.
// scores = ||c||^2 + (-2x)@c via bf16 WMMA / f32 accumulate; argmin over 512
// columns; exact-f32 epilogue (gather/ste/loss/dw/histogram).
// -----------------------------------------------------------------------------
__global__ __launch_bounds__(256) void vq_main(const float* __restrict__ flat,
                                               const float* __restrict__ cb,
                                               char* __restrict__ ws,
                                               float* __restrict__ ste) {
    __shared__ __align__(16) __bf16 s_cbfrag[32768];   // 64 KB

    const __bf16* g_cbfrag = (const __bf16*)(ws + WS_CBFRAG);
    const float*  g_cnorm  = (const float*)(ws + WS_CNORM);
    float*        hist     = (float*)(ws + WS_HIST);
    float*        lossa    = (float*)(ws + WS_LOSS);
    float*        dwacc    = (float*)(ws + WS_DWACC);

    // ---- stage codebook fragments into LDS once per workgroup (TDM DMA) ----
#if USE_TDM
    if (threadIdx.x < 32) {                 // one wave issues the tensor DMA
        unsigned ldsAddr = (unsigned)(size_t)(void*)&s_cbfrag[0];   // flat LDS addr low bits == LDS offset
        unsigned gLo = (unsigned)((size_t)g_cbfrag & 0xFFFFFFFFu);
        unsigned gHi = (unsigned)((size_t)g_cbfrag >> 32);
        u32x4 g0;
        g0[0] = 1u;                                     // count=1, user mode, no gather
        g0[1] = ldsAddr;                                // lds_addr
        g0[2] = gLo;                                    // global_addr[31:0]
        g0[3] = (gHi & 0x01FFFFFFu) | (2u << 30);       // global_addr[56:32] | type=2
        i32x8 g1;
        g1[0] = (3 << 16);                              // data_size=3 (8B), mask=0
        g1[1] = (int)(0x2000u << 16);                   // tensor_dim0[15:0]=8192 in [31:16]
        g1[2] = (int)(1u << 16);                        // tensor_dim0 hi=0 ; tensor_dim1 lo16=1
        g1[3] = (int)(0x2000u << 16);                   // tensor_dim1 hi=0 ; tile_dim0=8192
        g1[4] = 1;                                      // tile_dim1=1 ; tile_dim2=0
        g1[5] = 0x2000;                                 // tensor_dim0_stride[31:0]=8192
        g1[6] = 0;                                      // stride hi ; tensor_dim1_stride lo
        g1[7] = 0;
        i32x4 zz4; zz4[0] = 0; zz4[1] = 0; zz4[2] = 0; zz4[3] = 0;
        i32x8 zz8;
#pragma unroll
        for (int i = 0; i < 8; ++i) zz8[i] = 0;
        __builtin_amdgcn_tensor_load_to_lds(g0, g1, zz4, zz4, zz8, 0);
    }
    __builtin_amdgcn_s_wait_tensorcnt(0);
    __syncthreads();
#else
    {
        const uint4* src = (const uint4*)g_cbfrag;
        uint4*       dst = (uint4*)s_cbfrag;
        for (int i = threadIdx.x; i < 4096; i += 256) dst[i] = src[i];
    }
    __syncthreads();
#endif

    const int wave = threadIdx.x >> 5;
    const int lane = threadIdx.x & 31;
    const int half = lane >> 4;
    const int lm   = lane & 15;
    const int rowBase = (blockIdx.x * 8 + wave) * 32;   // 32 rows per wave

    // ---- build A fragments (bf16 of -2x) for two 16-row tiles ----
    // 16-bit A 16x32 layout: lane m (m<16): e0..7 -> K 0..7, e8..15 -> K 16..23;
    // lane m+16: e0..7 -> K 8..15, e8..15 -> K 24..31.
    auto buildA = [&](const float* xr, v16bf& f0, v16bf& f1) {
        float av[32];
#pragma unroll
        for (int c = 0; c < 4; ++c) {        // chunk c covers cols c*16 + half*8 ..+7
            float4 p0 = *(const float4*)(xr + c * 16 + half * 8);
            float4 p1 = *(const float4*)(xr + c * 16 + half * 8 + 4);
            av[c*8+0] = p0.x; av[c*8+1] = p0.y; av[c*8+2] = p0.z; av[c*8+3] = p0.w;
            av[c*8+4] = p1.x; av[c*8+5] = p1.y; av[c*8+6] = p1.z; av[c*8+7] = p1.w;
        }
#pragma unroll
        for (int e = 0; e < 8; ++e) {
            f0[e]     = (__bf16)(-2.f * av[e]);
            f0[e + 8] = (__bf16)(-2.f * av[8 + e]);
            f1[e]     = (__bf16)(-2.f * av[16 + e]);
            f1[e + 8] = (__bf16)(-2.f * av[24 + e]);
        }
    };
    v16bf a00, a01, a10, a11;
    buildA(flat + (size_t)(rowBase + lm) * DIM,      a00, a01);
    buildA(flat + (size_t)(rowBase + 16 + lm) * DIM, a10, a11);

    // ---- preload per-lane column norms into registers (no loads in hot loop) ----
    float cnr[NTILES];
#pragma unroll
    for (int t = 0; t < NTILES; ++t) cnr[t] = g_cnorm[t * 16 + lm];

    // ---- score GEMM + running argmin: 2 column tiles / iteration ----
    float minv[16];
    int   mini[16];
#pragma unroll
    for (int j = 0; j < 16; ++j) { minv[j] = 3.402823e38f; mini[j] = 0; }

    for (int t = 0; t < NTILES; t += 2) {
        // B fragments for column tiles t and t+1 (2 K-blocks each)
        const v16bf b00 = *(const v16bf*)(s_cbfrag + (((t + 0) * 2 + 0) * 32 + lane) * 16);
        const v16bf b01 = *(const v16bf*)(s_cbfrag + (((t + 0) * 2 + 1) * 32 + lane) * 16);
        const v16bf b10 = *(const v16bf*)(s_cbfrag + (((t + 1) * 2 + 0) * 32 + lane) * 16);
        const v16bf b11 = *(const v16bf*)(s_cbfrag + (((t + 1) * 2 + 1) * 32 + lane) * 16);
        v8f acc00, acc01, acc10, acc11;
#pragma unroll
        for (int j = 0; j < 8; ++j) {
            acc00[j] = cnr[t];     acc10[j] = cnr[t];       // rows 0-15 / 16-31, tile t
            acc01[j] = cnr[t + 1]; acc11[j] = cnr[t + 1];   // rows 0-15 / 16-31, tile t+1
        }
        // 4 independent accumulation chains; dependent pairs separated by 3 WMMAs
        acc00 = __builtin_amdgcn_wmma_f32_16x16x32_bf16(false, a00, false, b00,
                                                        (short)0, acc00, false, false);
        acc10 = __builtin_amdgcn_wmma_f32_16x16x32_bf16(false, a10, false, b00,
                                                        (short)0, acc10, false, false);
        acc01 = __builtin_amdgcn_wmma_f32_16x16x32_bf16(false, a00, false, b10,
                                                        (short)0, acc01, false, false);
        acc11 = __builtin_amdgcn_wmma_f32_16x16x32_bf16(false, a10, false, b10,
                                                        (short)0, acc11, false, false);
        acc00 = __builtin_amdgcn_wmma_f32_16x16x32_bf16(false, a01, false, b01,
                                                        (short)0, acc00, false, false);
        acc10 = __builtin_amdgcn_wmma_f32_16x16x32_bf16(false, a11, false, b01,
                                                        (short)0, acc10, false, false);
        acc01 = __builtin_amdgcn_wmma_f32_16x16x32_bf16(false, a01, false, b11,
                                                        (short)0, acc01, false, false);
        acc11 = __builtin_amdgcn_wmma_f32_16x16x32_bf16(false, a11, false, b11,
                                                        (short)0, acc11, false, false);
        const int col0 = (t + 0) * 16 + lm;
        const int col1 = (t + 1) * 16 + lm;
#pragma unroll
        for (int j = 0; j < 8; ++j) {
            float s;                          // lane<16 -> M=j, lane>=16 -> M=j+8
            s = acc00[j]; if (s < minv[j])     { minv[j]     = s; mini[j]     = col0; }
            s = acc01[j]; if (s < minv[j])     { minv[j]     = s; mini[j]     = col1; }
            s = acc10[j]; if (s < minv[8 + j]) { minv[8 + j] = s; mini[8 + j] = col0; }
            s = acc11[j]; if (s < minv[8 + j]) { minv[8 + j] = s; mini[8 + j] = col1; }
        }
    }

    // reduce across the 16 lanes sharing each row group (masks 1..8 stay in-half)
#pragma unroll
    for (int m = 1; m <= 8; m <<= 1) {
#pragma unroll
        for (int j = 0; j < 16; ++j) {
            float ov = __shfl_xor(minv[j], m, 32);
            int   oi = __shfl_xor(mini[j], m, 32);
            if (ov < minv[j] || (ov == minv[j] && oi < mini[j])) {
                minv[j] = ov; mini[j] = oi;
            }
        }
    }
    // set s (0/1): lanes 0..15 hold rows rowBase+s*16+0..7 (j), lanes 16..31 rows +8..15

    // ---- exact-f32 epilogue: gather, ste, loss, dw scatter, histogram ----
    float lossSum = 0.f;
    const int d0 = lane, d1 = lane + 32;
#pragma unroll
    for (int j2 = 0; j2 < 32; ++j2) {
        const int s  = j2 >> 4;
        const int jr = j2 & 15;
        int idx = __shfl(mini[s * 8 + (jr & 7)], (jr >> 3) * 16, 32);
        int row = rowBase + j2;
        float x0 = flat[(size_t)row * DIM + d0];
        float x1 = flat[(size_t)row * DIM + d1];
        float q0 = cb[d0 * KCODES + idx];
        float q1 = cb[d1 * KCODES + idx];
        ste[(size_t)row * DIM + d0] = x0 + (q0 - x0);   // mirrors reference STE fp ops
        ste[(size_t)row * DIM + d1] = x1 + (q1 - x1);
        float e0 = q0 - x0, e1 = q1 - x1;
        lossSum += e0 * e0 + e1 * e1;
        atomicAdd(&dwacc[d0 * KCODES + idx], x0);
        atomicAdd(&dwacc[d1 * KCODES + idx], x1);
        if (lane == 0) atomicAdd(&hist[idx], 1.0f);
    }
#pragma unroll
    for (int m = 16; m >= 1; m >>= 1) lossSum += __shfl_xor(lossSum, m, 32);
    if (lane == 0) atomicAdd(lossa, lossSum);
}

// -----------------------------------------------------------------------------
// Finalize: EMA update, bias correction, Laplace smoothing, perplexity, loss.
// One block of 512 threads (thread k owns code k).
// -----------------------------------------------------------------------------
__global__ __launch_bounds__(512) void vq_finalize(const float* __restrict__ ema_cs,
                                                   const float* __restrict__ ema_dw,
                                                   const int*   __restrict__ counter,
                                                   char* __restrict__ ws,
                                                   float* __restrict__ out) {
    __shared__ float red[512];
    __shared__ float s_n, s_ent;
    const float* hist  = (const float*)(ws + WS_HIST);
    const float* lossa = (const float*)(ws + WS_LOSS);
    const float* dwacc = (const float*)(ws + WS_DWACC);

    int k = threadIdx.x;
    int cnt = counter[0] + 1;
    float bias = 1.f - powf(GAMMA_, (float)cnt);

    float cs  = hist[k];
    float ncs = ema_cs[k] * GAMMA_ + cs * (1.f - GAMMA_);
    float avg = ncs / bias;

    // n = sum(avg_cs)
    red[k] = avg; __syncthreads();
    for (int s = 256; s > 0; s >>= 1) { if (k < s) red[k] += red[k + s]; __syncthreads(); }
    if (k == 0) s_n = red[0];
    __syncthreads();
    float n = s_n;

    // entropy for perplexity: p = cs / N
    float p = cs / (float)N_ROWS;
    red[k] = p * logf(p + 1e-10f); __syncthreads();
    for (int s = 256; s > 0; s >>= 1) { if (k < s) red[k] += red[k + s]; __syncthreads(); }
    if (k == 0) s_ent = red[0];
    __syncthreads();

    float smoothed = (avg + EPS_) / (n + (float)KCODES * EPS_) * n;

    out[O_CS + k] = ncs;
    for (int d = 0; d < DIM; ++d) {
        float dv = ema_dw[d * KCODES + k] * GAMMA_ + dwacc[d * KCODES + k] * (1.f - GAMMA_);
        out[O_DW + d * KCODES + k] = dv;
        out[O_CB + d * KCODES + k] = (dv / bias) / smoothed;
    }
    if (k == 0) {
        out[O_PERP] = expf(-s_ent);
        out[O_LOSS] = BETA_ * lossa[0] / (float)(N_ROWS * DIM);
    }
}

// -----------------------------------------------------------------------------
extern "C" void kernel_launch(void* const* d_in, const int* in_sizes, int n_in,
                              void* d_out, int out_size, void* d_ws, size_t ws_size,
                              hipStream_t stream) {
    (void)in_sizes; (void)n_in; (void)out_size; (void)ws_size;
    const float* flat   = (const float*)d_in[0];   // inputs  [64,32,32,64] -> [65536,64]
    const float* cb     = (const float*)d_in[1];   // codebook [64,512]
    const float* ema_cs = (const float*)d_in[2];   // [512]
    const float* ema_dw = (const float*)d_in[3];   // [64,512]
    const int*   counter = (const int*)d_in[4];    // scalar
    float* out = (float*)d_out;
    char*  ws  = (char*)d_ws;

    vq_prep<<<144, 256, 0, stream>>>(cb, ws);                 // 36864 threads
    vq_main<<<256, 256, 0, stream>>>(flat, cb, ws, out + O_STE);   // 32 rows / wave
    vq_finalize<<<1, 512, 0, stream>>>(ema_cs, ema_dw, counter, ws, out);
}
